// Unet_GNN_49924699849315
// MI455X (gfx1250) — matrix operand order repaired and checked by reading, not compiled, not run on hardware
//
#include <hip/hip_runtime.h>
#include <math.h>

typedef __attribute__((ext_vector_type(16))) _Float16 v16h;
typedef __attribute__((ext_vector_type(8)))  _Float16 v8h;
typedef __attribute__((ext_vector_type(8)))  float    v8f;

#define BN_EPS 1e-5f

// s_wait_asynccnt: builtin if available (probe: present on this toolchain), else asm.
#if __has_builtin(__builtin_amdgcn_s_wait_asynccnt)
#define WAIT_ASYNC(n) __builtin_amdgcn_s_wait_asynccnt(n)
#else
#define WAIT_ASYNC(n) asm volatile("s_wait_asynccnt %0" ::"i"(n))
#endif

// ---------------- encoder kernels ----------------

__global__ void conv3d_kernel(const float* __restrict__ in, const float* __restrict__ w,
                              const float* __restrict__ bias, float* __restrict__ out,
                              int B, int Ci, int Co, int D, int H, int W) {
  int idx = blockIdx.x * blockDim.x + threadIdx.x;
  int S = D * H * W;
  int total = B * Co * S;
  if (idx >= total) return;
  int x = idx % W; int t = idx / W;
  int y = t % H; t /= H;
  int z = t % D; t /= D;
  int co = t % Co; int b = t / Co;
  float acc = bias[co];
  const float* wbase = w + (size_t)co * Ci * 27;
  for (int ci = 0; ci < Ci; ++ci) {
    const float* ibase = in + ((size_t)(b * Ci + ci)) * S;
    const float* wc = wbase + ci * 27;
    for (int kd = 0; kd < 3; ++kd) {
      int zz = z + kd - 1; if (zz < 0 || zz >= D) continue;
      for (int kh = 0; kh < 3; ++kh) {
        int yy = y + kh - 1; if (yy < 0 || yy >= H) continue;
        for (int kw = 0; kw < 3; ++kw) {
          int xx = x + kw - 1; if (xx < 0 || xx >= W) continue;
          acc += ibase[(zz * H + yy) * W + xx] * wc[(kd * 3 + kh) * 3 + kw];
        }
      }
    }
  }
  out[idx] = acc;
}

__global__ void bn_stats_kernel(const float* __restrict__ x, float* __restrict__ mean,
                                float* __restrict__ inv, int B, int C, int S) {
  int c = blockIdx.x;
  __shared__ float s1[256], s2[256];
  float p1 = 0.f, p2 = 0.f;
  int cnt = B * S;
  for (int i = threadIdx.x; i < cnt; i += blockDim.x) {
    int b = i / S, sp = i % S;
    float v = x[((size_t)(b * C + c)) * S + sp];
    p1 += v; p2 += v * v;
  }
  s1[threadIdx.x] = p1; s2[threadIdx.x] = p2; __syncthreads();
  for (int st = 128; st > 0; st >>= 1) {
    if ((int)threadIdx.x < st) { s1[threadIdx.x] += s1[threadIdx.x + st]; s2[threadIdx.x] += s2[threadIdx.x + st]; }
    __syncthreads();
  }
  if (threadIdx.x == 0) {
    float m = s1[0] / cnt;
    float v = s2[0] / cnt - m * m;
    mean[c] = m;
    inv[c] = rsqrtf(v + BN_EPS);
  }
}

__global__ void bn_relu_kernel(float* __restrict__ x, const float* __restrict__ mean,
                               const float* __restrict__ inv, const float* __restrict__ g,
                               const float* __restrict__ be, int C, int S, int total) {
  int idx = blockIdx.x * blockDim.x + threadIdx.x;
  if (idx >= total) return;
  int c = (idx / S) % C;
  float v = (x[idx] - mean[c]) * inv[c] * g[c] + be[c];
  x[idx] = v > 0.f ? v : 0.f;
}

__global__ void maxpool_kernel(const float* __restrict__ in, float* __restrict__ out,
                               int BC, int D, int H, int W, int Do, int Ho, int Wo) {
  int idx = blockIdx.x * blockDim.x + threadIdx.x;
  int total = BC * Do * Ho * Wo;
  if (idx >= total) return;
  int x = idx % Wo; int t = idx / Wo;
  int y = t % Ho; t /= Ho;
  int z = t % Do; int bc = t / Do;
  const float* ib = in + (size_t)bc * D * H * W;
  float m = -INFINITY;
  for (int dz = 0; dz < 2; ++dz) {
    int zz = 2 * z - 1 + dz; if (zz < 0 || zz >= D) continue;
    for (int dy = 0; dy < 2; ++dy) {
      int yy = 2 * y - 1 + dy; if (yy < 0 || yy >= H) continue;
      for (int dx = 0; dx < 2; ++dx) {
        int xx = 2 * x - 1 + dx; if (xx < 0 || xx >= W) continue;
        float v = ib[(zz * H + yy) * W + xx];
        m = v > m ? v : m;
      }
    }
  }
  out[idx] = m;
}

// ---------------- conversion / utility kernels ----------------

__global__ void convert_strided_f16(const float* __restrict__ src, _Float16* __restrict__ dst,
                                    int rows, int cols, int ld, int coff) {
  size_t idx = (size_t)blockIdx.x * blockDim.x + threadIdx.x;
  size_t total = (size_t)rows * cols;
  if (idx >= total) return;
  size_t r = idx / cols, c = idx % cols;
  dst[idx] = (_Float16)src[r * ld + coff + c];
}

__global__ void f32_to_f16_kernel(const float* __restrict__ s, _Float16* __restrict__ d, size_t n) {
  size_t i = (size_t)blockIdx.x * blockDim.x + threadIdx.x;
  if (i < n) d[i] = (_Float16)s[i];
}

__global__ void pad_enc_kernel(const float* __restrict__ enc, _Float16* __restrict__ out) {
  int idx = blockIdx.x * blockDim.x + threadIdx.x;
  if (idx >= 16 * 12288) return;
  int r = idx / 12288;
  out[idx] = (r < 4) ? (_Float16)enc[idx] : (_Float16)0.0f;
}

__global__ void fill_f32(float* p, float v, int n) {
  int i = blockIdx.x * blockDim.x + threadIdx.x;
  if (i < n) p[i] = v;
}

__global__ void fill_u32(unsigned* p, unsigned v, int n) {
  int i = blockIdx.x * blockDim.x + threadIdx.x;
  if (i < n) p[i] = v;
}

// ---------------- WMMA GEMM (direct-load variant, used for small M) ----------------
// C[M,N] = A[M,K] * Bw[N,K]^T, f16 in / f32 out. Wave computes 16x64.
__global__ void wmma_gemm_f16_direct(const _Float16* __restrict__ A, const _Float16* __restrict__ Bw,
                                     float* __restrict__ C, int M, int N, int K) {
  int lane = threadIdx.x & 31;
  int wave = threadIdx.x >> 5;
  int tilesN = N >> 6;
  int tilesM = M >> 4;
  int tile = blockIdx.x * 4 + wave;
  if (tile >= tilesM * tilesN) return;
  int tm = (tile / tilesN) << 4;
  int tn = (tile % tilesN) << 6;
  int m  = lane & 15;
  int hi = lane >> 4;

  v8f acc[4] = {};
  const _Float16* arow = A + (size_t)(tm + m) * K;

  for (int k0 = 0; k0 < K; k0 += 32) {
    union { v16h v; v8h h[2]; } af;
    af.h[0] = *(const v8h*)(arow + k0 + hi * 16);
    af.h[1] = *(const v8h*)(arow + k0 + hi * 16 + 8);
#pragma unroll
    for (int j = 0; j < 4; ++j) {
      const _Float16* br = Bw + (size_t)(tn + j * 16 + m) * K + k0 + hi * 16;
      v16h bf = *(const v16h*)br;
      acc[j] = __builtin_amdgcn_wmma_f32_16x16x32_f16(false, af.v, false, bf,
                                                      (short)0, acc[j], false, false);
    }
  }

#pragma unroll
  for (int j = 0; j < 4; ++j) {
    float* crow = C + (size_t)(tm + hi * 8) * N + tn + j * 16 + m;
#pragma unroll
    for (int r = 0; r < 8; ++r) crow[(size_t)r * N] = acc[j][r];
  }
}

// ---------------- WMMA GEMM with LDS-staged B + async copy double buffering ----------
// Block = 128 threads (4 waves). Block computes 128(M) x 64(N); each wave 32x64
// (two 16-row tiles sharing the B fragments). B tile (64 rows x 32 k, f16) staged
// in LDS once per block per k-step via GLOBAL_LOAD_ASYNC_TO_LDS_B128 (inline asm,
// double buffered on ASYNCcnt), rows padded to 40 halves for bank spreading.
// Requires M % 128 == 0 && N % 64 == 0 (grid exactly covers C; no early returns).
#define BROW 40  // padded LDS row stride in halves

__device__ __forceinline__ void stage_b_tile(const _Float16* __restrict__ Bw, _Float16* bs,
                                             int tn, int k0, int K, int tid) {
#pragma unroll
  for (int c = tid; c < 256; c += 128) {       // 256 chunks of 16B = 64 rows x 32 halves
    int n_local = c >> 2;
    int koff = (c & 3) * 8;
    const _Float16* g = Bw + (size_t)(tn + n_local) * K + k0 + koff;
    _Float16* l = bs + n_local * BROW + koff;
#if defined(__gfx1250__)
    // VDST = per-lane LDS byte address (low 32 bits of the generic address of a
    // __shared__ object are the LDS offset, ISA 10.2), VADDR = 64-bit global addr.
    unsigned lofs = (unsigned)(size_t)l;
    unsigned long long ga = (unsigned long long)(size_t)g;
    asm volatile("global_load_async_to_lds_b128 %0, %1, off"
                 :
                 : "v"(lofs), "v"(ga)
                 : "memory");
#else
    *(v8h*)l = *(const v8h*)g;
#endif
  }
}

__global__ void wmma_gemm_f16_lds(const _Float16* __restrict__ A, const _Float16* __restrict__ Bw,
                                  float* __restrict__ C, int M, int N, int K) {
  __shared__ _Float16 Bs[2][64 * BROW];
  int tid  = threadIdx.x;
  int lane = tid & 31;
  int wave = tid >> 5;
  int tilesN = N >> 6;
  int tn = (blockIdx.x % tilesN) << 6;
  int tmBlock = (blockIdx.x / tilesN) << 7;   // 128 rows per block
  int tmw = tmBlock + wave * 32;              // this wave's 32 rows
  int m  = lane & 15;
  int hi = lane >> 4;

  const _Float16* arow0 = A + (size_t)(tmw + m) * K;        // tile 0 rows
  const _Float16* arow1 = A + (size_t)(tmw + 16 + m) * K;   // tile 1 rows

  v8f acc0[4] = {};
  v8f acc1[4] = {};

  stage_b_tile(Bw, &Bs[0][0], tn, 0, K, tid);
  int cur = 0;

  for (int k0 = 0; k0 < K; k0 += 32) {
    if (k0 + 32 < K) {
      stage_b_tile(Bw, &Bs[cur ^ 1][0], tn, k0 + 32, K, tid);
#if defined(__gfx1250__)
      WAIT_ASYNC(2);   // previous tile's 2 per-wave async loads done (in-order)
#endif
    } else {
#if defined(__gfx1250__)
      WAIT_ASYNC(0);
#endif
    }
    __syncthreads();

    if (k0 + 128 < K) {                      // stream A ahead (global_prefetch_b8)
      __builtin_prefetch(arow0 + k0 + 128, 0, 1);
      __builtin_prefetch(arow1 + k0 + 128, 0, 1);
    }

    union { v16h v; v8h h[2]; } a0, a1;
    a0.h[0] = *(const v8h*)(arow0 + k0 + hi * 16);
    a0.h[1] = *(const v8h*)(arow0 + k0 + hi * 16 + 8);
    a1.h[0] = *(const v8h*)(arow1 + k0 + hi * 16);
    a1.h[1] = *(const v8h*)(arow1 + k0 + hi * 16 + 8);

    const _Float16* bsCur = &Bs[cur][0];
#pragma unroll
    for (int j = 0; j < 4; ++j) {
      union { v16h v; v8h h[2]; } bf;
      const _Float16* br = bsCur + (j * 16 + m) * BROW + hi * 16;
      bf.h[0] = *(const v8h*)(br);
      bf.h[1] = *(const v8h*)(br + 8);
      acc0[j] = __builtin_amdgcn_wmma_f32_16x16x32_f16(false, a0.v, false, bf.v,
                                                       (short)0, acc0[j], false, false);
      acc1[j] = __builtin_amdgcn_wmma_f32_16x16x32_f16(false, a1.v, false, bf.v,
                                                       (short)0, acc1[j], false, false);
    }
    __syncthreads();
    cur ^= 1;
  }

#pragma unroll
  for (int j = 0; j < 4; ++j) {
    float* c0 = C + (size_t)(tmw + hi * 8) * N + tn + j * 16 + m;
    float* c1 = C + (size_t)(tmw + 16 + hi * 8) * N + tn + j * 16 + m;
#pragma unroll
    for (int r = 0; r < 8; ++r) {
      c0[(size_t)r * N] = acc0[j][r];
      c1[(size_t)r * N] = acc1[j][r];
    }
  }
}

// ---------------- GAT kernels ----------------

__global__ void build_h0_kernel(const float* __restrict__ henc, const float* __restrict__ xyz,
                                const int* __restrict__ graph, const float* __restrict__ W0,
                                float* __restrict__ h, int N) {
  int idx = blockIdx.x * blockDim.x + threadIdx.x;
  if (idx >= N * 1024) return;
  int n = idx >> 10, o = idx & 1023;
  int g = graph[n];
  const float* wr = W0 + (size_t)o * 12291;
  h[idx] = henc[g * 1024 + o]
         + xyz[n * 3 + 0] * wr[0] + xyz[n * 3 + 1] * wr[1] + xyz[n * 3 + 2] * wr[2];
}

__global__ void matvec_kernel(const float* __restrict__ x, const float* __restrict__ w,
                              float* __restrict__ out, int K) {
  int n = blockIdx.x;
  __shared__ float s[256];
  float p = 0.f;
  for (int k = threadIdx.x; k < K; k += blockDim.x) p += x[(size_t)n * K + k] * w[k];
  s[threadIdx.x] = p; __syncthreads();
  for (int st = 128; st > 0; st >>= 1) {
    if ((int)threadIdx.x < st) s[threadIdx.x] += s[threadIdx.x + st];
    __syncthreads();
  }
  if (threadIdx.x == 0) out[n] = s[0];
}

__global__ void gat_scores(const float* __restrict__ h, const float* __restrict__ a_src,
                           const float* __restrict__ a_dst, float* __restrict__ s_src,
                           float* __restrict__ s_dst, int fout) {
  int n = blockIdx.x;
  __shared__ float r1[256], r2[256];
  float p1 = 0.f, p2 = 0.f;
  for (int k = threadIdx.x; k < fout; k += blockDim.x) {
    float v = h[(size_t)n * fout + k];
    p1 += v * a_src[k];
    p2 += v * a_dst[k];
  }
  r1[threadIdx.x] = p1; r2[threadIdx.x] = p2; __syncthreads();
  for (int st = 128; st > 0; st >>= 1) {
    if ((int)threadIdx.x < st) { r1[threadIdx.x] += r1[threadIdx.x + st]; r2[threadIdx.x] += r2[threadIdx.x + st]; }
    __syncthreads();
  }
  if (threadIdx.x == 0) { s_src[n] = r1[0]; s_dst[n] = r2[0]; }
}

__device__ inline unsigned f2o(float f) {
  unsigned b = __float_as_uint(f);
  return (b & 0x80000000u) ? ~b : (b | 0x80000000u);
}
__device__ inline float o2f(unsigned u) {
  return (u & 0x80000000u) ? __uint_as_float(u & 0x7fffffffu) : __uint_as_float(~u);
}

__global__ void edge_scores(const float* __restrict__ s_src, const float* __restrict__ s_dst,
                            const int* __restrict__ srcE, const int* __restrict__ dstE,
                            int E0, int ET, float* __restrict__ ebuf, unsigned* __restrict__ emax) {
  int e = blockIdx.x * blockDim.x + threadIdx.x;
  if (e >= ET) return;
  int s = e < E0 ? srcE[e] : (e - E0);
  int d = e < E0 ? dstE[e] : (e - E0);
  float v = s_src[s] + s_dst[d];
  v = v > 0.f ? v : 0.2f * v;  // leaky_relu(0.2)
  ebuf[e] = v;
  atomicMax(&emax[d], f2o(v));
}

__global__ void edge_exp(const float* __restrict__ ebuf, const unsigned* __restrict__ emax,
                         const int* __restrict__ dstE, int E0, int ET,
                         float* __restrict__ exbuf, float* __restrict__ denom, float* __restrict__ deg) {
  int e = blockIdx.x * blockDim.x + threadIdx.x;
  if (e >= ET) return;
  int d = e < E0 ? dstE[e] : (e - E0);
  float ex = expf(ebuf[e] - o2f(emax[d]));
  exbuf[e] = ex;
  atomicAdd(&denom[d], ex);
  atomicAdd(&deg[d], 1.0f);
}

__global__ void gat_aggregate(const float* __restrict__ h, const float* __restrict__ exbuf,
                              const float* __restrict__ denom, const int* __restrict__ srcE,
                              const int* __restrict__ dstE, int E0, float* __restrict__ agg, int fout) {
  int e = blockIdx.x;
  int s = e < E0 ? srcE[e] : (e - E0);
  int d = e < E0 ? dstE[e] : (e - E0);
  float alpha = exbuf[e] / denom[d];
  for (int o = threadIdx.x; o < fout; o += blockDim.x)
    atomicAdd(&agg[(size_t)d * fout + o], alpha * h[(size_t)s * fout + o]);
}

__global__ void gat_finalize(float* __restrict__ agg, const float* __restrict__ deg,
                             const float* __restrict__ bias, int N, int fout) {
  int i = blockIdx.x * blockDim.x + threadIdx.x;
  if (i >= N * fout) return;
  int n = i / fout, o = i % fout;
  agg[i] = agg[i] / deg[n] + bias[o];
}

__global__ void softplus_kernel(const float* __restrict__ x, float* __restrict__ out, int n) {
  int i = blockIdx.x * blockDim.x + threadIdx.x;
  if (i < n) out[i] = logf(expf(x[i]) + 1.0f);
}

// ---------------- host launch ----------------

extern "C" void kernel_launch(void* const* d_in, const int* in_sizes, int n_in,
                              void* d_out, int out_size, void* d_ws, size_t ws_size,
                              hipStream_t stream) {
  (void)n_in; (void)out_size; (void)ws_size;
  const float* X     = (const float*)d_in[0];
  const float* xyz   = (const float*)d_in[1];
  const int*   graph = (const int*)d_in[2];
  const int*   edge  = (const int*)d_in[3];
  int E0 = in_sizes[3] / 2;     // 65536
  const int* srcE = edge;
  const int* dstE = edge + E0;
  int N  = in_sizes[2];         // 4096
  int ET = E0 + N;              // edges + self loops

  // --- workspace carve ---
  char* base = (char*)d_ws; size_t off = 0;
  auto alloc = [&](size_t bytes) -> char* {
    off = (off + 255) & ~(size_t)255;
    char* r = base + off; off += bytes; return r;
  };
  float*    bufX  = (float*)alloc((size_t)460800 * 4);
  float*    bufY  = (float*)alloc((size_t)460800 * 4);
  float*    bufZ  = (float*)alloc((size_t)460800 * 4);
  float*    mean  = (float*)alloc(1024 * 4);
  float*    invs  = (float*)alloc(1024 * 4);
  _Float16* enc_h = (_Float16*)alloc((size_t)16 * 12288 * 2);
  _Float16* Wb_h  = (_Float16*)alloc((size_t)1024 * 12288 * 2); // reused per layer
  _Float16* x_h   = (_Float16*)alloc((size_t)4096 * 1024 * 2);
  float*    henc  = (float*)alloc((size_t)16 * 1024 * 4);
  float*    hbuf  = (float*)alloc((size_t)4096 * 1024 * 4);
  float*    xA    = (float*)alloc((size_t)4096 * 1024 * 4);
  float*    xB    = (float*)alloc((size_t)4096 * 1024 * 4);
  float*    ssrc  = (float*)alloc((size_t)N * 4);
  float*    sdst  = (float*)alloc((size_t)N * 4);
  unsigned* emax  = (unsigned*)alloc((size_t)N * 4);
  float*    denom = (float*)alloc((size_t)N * 4);
  float*    deg   = (float*)alloc((size_t)N * 4);
  float*    ebuf  = (float*)alloc((size_t)ET * 4);
  float*    exbuf = (float*)alloc((size_t)ET * 4);

  // --- encoder ---
  int ch[6]  = {25, 64, 128, 256, 512, 1024};
  int dd[5]  = {18, 10, 6, 4, 3};
  int hh2[5] = {10, 6, 4, 3, 2};
  int ww2[5] = {10, 6, 4, 3, 2};
  const float* cur = X;
  for (int i = 0; i < 5; ++i) {
    int Ci = ch[i], Co = ch[i + 1];
    int D = dd[i], H = hh2[i], Wd = ww2[i];
    int S = D * H * Wd;
    const float* w1  = (const float*)d_in[4 + 8 * i];
    const float* b1  = (const float*)d_in[5 + 8 * i];
    const float* g1  = (const float*)d_in[6 + 8 * i];
    const float* be1 = (const float*)d_in[7 + 8 * i];
    const float* w2  = (const float*)d_in[8 + 8 * i];
    const float* b2  = (const float*)d_in[9 + 8 * i];
    const float* g2  = (const float*)d_in[10 + 8 * i];
    const float* be2 = (const float*)d_in[11 + 8 * i];
    int tot1 = 4 * Co * S;
    conv3d_kernel<<<(tot1 + 255) / 256, 256, 0, stream>>>(cur, w1, b1, bufX, 4, Ci, Co, D, H, Wd);
    bn_stats_kernel<<<Co, 256, 0, stream>>>(bufX, mean, invs, 4, Co, S);
    bn_relu_kernel<<<(tot1 + 255) / 256, 256, 0, stream>>>(bufX, mean, invs, g1, be1, Co, S, tot1);
    conv3d_kernel<<<(tot1 + 255) / 256, 256, 0, stream>>>(bufX, w2, b2, bufY, 4, Co, Co, D, H, Wd);
    bn_stats_kernel<<<Co, 256, 0, stream>>>(bufY, mean, invs, 4, Co, S);
    bn_relu_kernel<<<(tot1 + 255) / 256, 256, 0, stream>>>(bufY, mean, invs, g2, be2, Co, S, tot1);
    if (i < 4) {
      int Do = dd[i + 1], Ho = hh2[i + 1], Wo = ww2[i + 1];
      int tot2 = 4 * Co * Do * Ho * Wo;
      maxpool_kernel<<<(tot2 + 255) / 256, 256, 0, stream>>>(bufY, bufZ, 4 * Co, D, H, Wd, Do, Ho, Wo);
      cur = bufZ;
    } else {
      cur = bufY;
    }
  }
  const float* enc = cur;  // [4, 12288]

  // --- GAT layer 0: exploit enc[node_graph] being per-graph constant ---
  const float* W0 = (const float*)d_in[44];
  {
    size_t totw = (size_t)1024 * 12288;
    convert_strided_f16<<<(int)((totw + 255) / 256), 256, 0, stream>>>(W0, Wb_h, 1024, 12288, 12291, 3);
    pad_enc_kernel<<<(16 * 12288 + 255) / 256, 256, 0, stream>>>(enc, enc_h);
    int tiles = (16 / 16) * (1024 / 64);
    wmma_gemm_f16_direct<<<(tiles + 3) / 4, 128, 0, stream>>>(enc_h, Wb_h, henc, 16, 1024, 12288);
    build_h0_kernel<<<(N * 1024 + 255) / 256, 256, 0, stream>>>(henc, xyz, graph, W0, hbuf, N);
  }

  float* x_cur = xA;
  float* x_nxt = xB;
  for (int j = 0; j < 7; ++j) {
    int fout = (j == 6) ? 1 : 1024;
    const float* Wj  = (const float*)d_in[44 + 4 * j];
    const float* a_s = (const float*)d_in[45 + 4 * j];
    const float* a_d = (const float*)d_in[46 + 4 * j];
    const float* bj  = (const float*)d_in[47 + 4 * j];

    if (j == 0) {
      // hbuf already holds h for layer 0
    } else if (j < 6) {
      size_t totc = (size_t)N * 1024;
      f32_to_f16_kernel<<<(int)((totc + 255) / 256), 256, 0, stream>>>(x_cur, x_h, totc);
      size_t totw = (size_t)1024 * 1024;
      convert_strided_f16<<<(int)((totw + 255) / 256), 256, 0, stream>>>(Wj, Wb_h, 1024, 1024, 1024, 0);
      int blocks = (N / 128) * (1024 / 64);   // 128x64 per block
      wmma_gemm_f16_lds<<<blocks, 128, 0, stream>>>(x_h, Wb_h, hbuf, N, 1024, 1024);
    } else {
      matvec_kernel<<<N, 256, 0, stream>>>(x_cur, Wj, hbuf, 1024);
    }

    // attention + mean aggregation
    gat_scores<<<N, 256, 0, stream>>>(hbuf, a_s, a_d, ssrc, sdst, fout);
    fill_u32<<<(N + 255) / 256, 256, 0, stream>>>(emax, 0x007FFFFFu, N);  // f2o(-inf)
    fill_f32<<<(N + 255) / 256, 256, 0, stream>>>(denom, 0.f, N);
    fill_f32<<<(N + 255) / 256, 256, 0, stream>>>(deg, 0.f, N);
    int aggtot = N * fout;
    fill_f32<<<(aggtot + 255) / 256, 256, 0, stream>>>(x_nxt, 0.f, aggtot);
    edge_scores<<<(ET + 255) / 256, 256, 0, stream>>>(ssrc, sdst, srcE, dstE, E0, ET, ebuf, emax);
    edge_exp<<<(ET + 255) / 256, 256, 0, stream>>>(ebuf, emax, dstE, E0, ET, exbuf, denom, deg);
    gat_aggregate<<<ET, 256, 0, stream>>>(hbuf, exbuf, denom, srcE, dstE, E0, x_nxt, fout);
    gat_finalize<<<(aggtot + 255) / 256, 256, 0, stream>>>(x_nxt, deg, bj, N, fout);

    float* t = x_cur; x_cur = x_nxt; x_nxt = t;
  }

  softplus_kernel<<<(N + 255) / 256, 256, 0, stream>>>(x_cur, (float*)d_out, N);
}
